// NONA_NN_54915451847256
// MI455X (gfx1250) — matrix operand order repaired and checked by reading, not compiled, not run on hardware
//
#include <hip/hip_runtime.h>

typedef __attribute__((ext_vector_type(16))) _Float16 v16h;
typedef __attribute__((ext_vector_type(8)))  _Float16 v8h;
typedef __attribute__((ext_vector_type(8)))  float    v8f;

// ---------------- WMMA tile loaders (16x16x32 f16, wave32) ----------------
// A tile: 16 rows (M) x 32 (K) from row-major [* x ld] at (r0, k0).
// ISA layout: lanes 0-15 (row m): elems 0-7 = K0..7, elems 8-15 = K16..23;
//             lanes 16-31:        elems 0-7 = K8..15, elems 8-15 = K24..31.
__device__ __forceinline__ v16h load_a16(const _Float16* Mp, int ld, int r0, int k0, int lane) {
  const _Float16* p = Mp + (size_t)(r0 + (lane & 15)) * (size_t)ld
                         + (size_t)(k0 + ((lane >> 4) << 3));
  v8h lo = *(const v8h*)p;
  v8h hi = *(const v8h*)(p + 16);
  v16h r;
#pragma unroll
  for (int i = 0; i < 8; ++i) { r[i] = lo[i]; r[i + 8] = hi[i]; }
  return r;
}

// B tile: 32 (K) x 16 (N) where B[k][n] = Mp[n0+n][k] (row-major source, rows = N dim).
// Layout: lanes 0-15 (col n): elems 0-7 = K0..7, elems 8-15 = K16..23; lanes 16-31: elems 0-7 = K8..15, elems 8-15 = K24..31.
__device__ __forceinline__ v16h load_b16(const _Float16* Mp, int ld, int n0, int k0, int lane) {
  const _Float16* p = Mp + (size_t)(n0 + (lane & 15)) * (size_t)ld
                         + (size_t)(k0 + ((lane >> 4) << 3));
  v8h lo = *(const v8h*)p;
  v8h hi = *(const v8h*)(p + 16);
  v16h r;
#pragma unroll
  for (int i = 0; i < 8; ++i) { r[i] = lo[i]; r[i + 8] = hi[i]; }
  return r;
}

__device__ __forceinline__ v8f wmma16(v16h a, v16h b, v8f c) {
  return __builtin_amdgcn_wmma_f32_16x16x32_f16(false, a, false, b, (short)0, c, false, false);
}

// ---------------- elementwise f32 -> f16 ----------------
__global__ void k_cvt_f16(const float* __restrict__ src, _Float16* __restrict__ dst, int n) {
  int i = blockIdx.x * blockDim.x + threadIdx.x;
  if (i < n) dst[i] = (_Float16)src[i];
}

// ---------------- transposed one-hot: Yt[c][j] = (y[j]==c), c in [0,16) ----------------
__global__ void k_onehot_t(const int* __restrict__ y, _Float16* __restrict__ Yt, int M) {
  int i = blockIdx.x * blockDim.x + threadIdx.x;
  if (i < 16 * M) {
    int c = i / M, j = i - c * M;
    Yt[i] = (_Float16)((y[j] == c) ? 1.0f : 0.0f);
  }
}

// ---------------- column batch-stats -> fused scale/shift: s=g*rsqrt(v+eps), t=b-m*s ----
template <typename T>
__global__ void k_colstats(const T* __restrict__ X, const float* __restrict__ gamma,
                           const float* __restrict__ beta, float* __restrict__ svec,
                           float* __restrict__ tvec, int N, int D) {
  int c = blockIdx.x * 64 + (threadIdx.x & 63);
  int w = threadIdx.x >> 6;  // 4 row-walkers x 64 cols
  float s = 0.f, ss = 0.f;
  for (int r = w; r < N; r += 4) {
    float v = (float)X[(size_t)r * D + c];
    s += v; ss += v * v;
  }
  __shared__ float ls[4][64], lq[4][64];
  ls[w][threadIdx.x & 63] = s;
  lq[w][threadIdx.x & 63] = ss;
  __syncthreads();
  if (w == 0) {
    for (int i = 1; i < 4; ++i) { s += ls[i][threadIdx.x & 63]; ss += lq[i][threadIdx.x & 63]; }
    float mean = s / (float)N;
    float var  = ss / (float)N - mean * mean;
    float sc   = gamma[c] * rsqrtf(var + 1e-5f);
    svec[c] = sc;
    tvec[c] = beta[c] - mean * sc;
  }
}

// ---------------- fold BN into weights: Wp = f16(W*s), bp = b + W@t ----------------
__global__ void k_wfold(const float* __restrict__ W, const float* __restrict__ bvec,
                        const float* __restrict__ svec, const float* __restrict__ tvec,
                        _Float16* __restrict__ Wp, float* __restrict__ bp, int din) {
  int o = blockIdx.x;
  float acc = 0.f;
  for (int c = threadIdx.x; c < din; c += blockDim.x) {
    float wv = W[(size_t)o * din + c];
    acc += wv * tvec[c];
    Wp[(size_t)o * din + c] = (_Float16)(wv * svec[c]);
  }
  __shared__ float red[256];
  red[threadIdx.x] = acc;
  __syncthreads();
  for (int s = 128; s > 0; s >>= 1) {
    if (threadIdx.x < s) red[threadIdx.x] += red[threadIdx.x + s];
    __syncthreads();
  }
  if (threadIdx.x == 0) bp[o] = bvec[o] + red[0];
}

// ---------------- H = tanh(A @ Wp^T + bp), A:[N,din] f16, Wp:[dout,din] f16 ----------------
// 32-row blocks; A block staged into LDS once via GLOBAL_LOAD_ASYNC_TO_LDS (ASYNCcnt),
// then shared by all 8 waves. Each wave computes 32x32 output tiles (4 WMMAs / k-step).
__global__ void __launch_bounds__(256) k_gemm_tanh(const _Float16* __restrict__ A,
    const _Float16* __restrict__ Wp, const float* __restrict__ bp,
    _Float16* __restrict__ H, int din, int dout) {
  extern __shared__ __align__(16) _Float16 As[];  // [32][din]
  int tid = threadIdx.x;
  int wave = tid >> 5, lane = tid & 31;
  int r0 = blockIdx.x * 32;

  {  // async-stage A[32][din] into LDS; generic shared ptr low 32 bits = LDS byte offset
    uint32_t lbase = (uint32_t)(uintptr_t)&As[0];
    uint64_t gbase = (uint64_t)(uintptr_t)(A + (size_t)r0 * din);
    int nvec = (32 * din) >> 3;  // number of b128 transfers
    for (int v = tid; v < nvec; v += 256) {
      uint32_t laddr = lbase + ((uint32_t)v << 4);
      uint64_t gaddr = gbase + ((uint64_t)v << 4);
      asm volatile("global_load_async_to_lds_b128 %0, %1, off"
                   :: "v"(laddr), "v"(gaddr) : "memory");
    }
    asm volatile("s_wait_asynccnt 0x0" ::: "memory");
    __syncthreads();
  }

  const _Float16* Al = As;     // read A tiles back via ds_load
  int ngrp = dout >> 5;        // 32-column groups
  for (int cg = wave; cg < ngrp; cg += 8) {
    v8f cc[2][2] = {};
    int c0 = cg << 5;
    for (int k0 = 0; k0 < din; k0 += 32) {
      v16h a0 = load_a16(Al, din, 0,  k0, lane);
      v16h a1 = load_a16(Al, din, 16, k0, lane);
      v16h b0 = load_b16(Wp, din, c0,      k0, lane);
      v16h b1 = load_b16(Wp, din, c0 + 16, k0, lane);
      cc[0][0] = wmma16(a0, b0, cc[0][0]);
      cc[0][1] = wmma16(a0, b1, cc[0][1]);
      cc[1][0] = wmma16(a1, b0, cc[1][0]);
      cc[1][1] = wmma16(a1, b1, cc[1][1]);
    }
    int rbl = (lane >> 4) << 3;
#pragma unroll
    for (int tj = 0; tj < 2; ++tj) {
      int col = c0 + tj * 16 + (lane & 15);
      float bvv = bp[col];
#pragma unroll
      for (int ti = 0; ti < 2; ++ti) {
#pragma unroll
        for (int v = 0; v < 8; ++v) {
          H[(size_t)(r0 + ti * 16 + rbl + v) * dout + col] =
              (_Float16)tanhf(cc[ti][tj][v] + bvv);
        }
      }
    }
  }
}

// ---------------- kk[j] = |k_j|^2 ----------------
__global__ void k_rownorm2(const _Float16* __restrict__ K, float* __restrict__ kk, int M, int D) {
  int j = blockIdx.x * blockDim.x + threadIdx.x;
  if (j < M) {
    float s = 0.f;
    for (int c = 0; c < D; ++c) { float v = (float)K[(size_t)j * D + c]; s += v * v; }
    kk[j] = s;
  }
}

// ---------------- pass 1: rmax[i] = max_j (2 q_i.k_j - |k_j|^2) ----------------
__global__ void __launch_bounds__(256) k_rowmax(const _Float16* __restrict__ Q,
    const _Float16* __restrict__ K, const float* __restrict__ kk,
    float* __restrict__ rmax, int M) {
  int wave = threadIdx.x >> 5, lane = threadIdx.x & 31;
  int r0 = blockIdx.x * 16;
  v16h qa[4];
#pragma unroll
  for (int i = 0; i < 4; ++i) qa[i] = load_a16(Q, 128, r0, i * 32, lane);
  int chunk = M >> 3;
  int j0 = wave * chunk, j1 = j0 + chunk;
  float mx[8];
#pragma unroll
  for (int v = 0; v < 8; ++v) mx[v] = -3.0e38f;
  for (int j = j0; j < j1; j += 16) {
    if (j + 16 < j1) __builtin_prefetch(K + (size_t)(j + 16) * 128, 0, 1);
    v8f c = {};
#pragma unroll
    for (int i = 0; i < 4; ++i) c = wmma16(qa[i], load_b16(K, 128, j, i * 32, lane), c);
    float kv = kk[j + (lane & 15)];
#pragma unroll
    for (int v = 0; v < 8; ++v) mx[v] = fmaxf(mx[v], 2.f * c[v] - kv);
  }
  // row max across the 16 lanes of each half-wave (rows differ between halves)
#pragma unroll
  for (int m = 1; m < 16; m <<= 1)
#pragma unroll
    for (int v = 0; v < 8; ++v) mx[v] = fmaxf(mx[v], __shfl_xor(mx[v], m, 32));
  __shared__ float lmax[8][16];
  if ((lane & 15) == 0) {
    int rb = (lane >> 4) << 3;
#pragma unroll
    for (int v = 0; v < 8; ++v) lmax[wave][rb + v] = mx[v];
  }
  __syncthreads();
  if (threadIdx.x < 16) {
    float m = lmax[0][threadIdx.x];
    for (int w = 1; w < 8; ++w) m = fmaxf(m, lmax[w][threadIdx.x]);
    rmax[r0 + threadIdx.x] = m;
  }
}

// ---------------- pass 2: out = clip(softmax(sim) @ onehot, 0, 1) ----------------
__global__ void __launch_bounds__(256) k_softmax_agg(const _Float16* __restrict__ Q,
    const _Float16* __restrict__ K, const float* __restrict__ kk,
    const _Float16* __restrict__ Yt, const float* __restrict__ rmax,
    float* __restrict__ out, int M) {
  int wave = threadIdx.x >> 5, lane = threadIdx.x & 31;
  int r0 = blockIdx.x * 16;
  int rb = (lane >> 4) << 3;
  v16h qa[4];
#pragma unroll
  for (int i = 0; i < 4; ++i) qa[i] = load_a16(Q, 128, r0, i * 32, lane);
  float rm[8];
#pragma unroll
  for (int v = 0; v < 8; ++v) rm[v] = rmax[r0 + rb + v];

  __shared__ __align__(32) _Float16 etile[8][16 * 32];  // per-wave 16x32 exp tile
  __shared__ float cacc[8][16][16];

  v8f acc = {};
  int chunk = M >> 3;
  int j0 = wave * chunk, j1 = j0 + chunk;
  for (int j = j0; j < j1; j += 32) {
    if (j + 32 < j1) __builtin_prefetch(K + (size_t)(j + 32) * 128, 0, 1);
#pragma unroll
    for (int t = 0; t < 2; ++t) {
      v8f c = {};
#pragma unroll
      for (int i = 0; i < 4; ++i)
        c = wmma16(qa[i], load_b16(K, 128, j + t * 16, i * 32, lane), c);
      float kv = kk[j + t * 16 + (lane & 15)];
#pragma unroll
      for (int v = 0; v < 8; ++v) {
        float e = __expf(2.f * c[v] - kv - rm[v]);  // <= 1 by construction
        etile[wave][(rb + v) * 32 + t * 16 + (lane & 15)] = (_Float16)e;
      }
    }
    // same-wave LDS RAW: DS ops execute in order; block compiler reordering + wait.
    asm volatile("s_wait_dscnt 0" ::: "memory");
    v16h ea = load_a16(&etile[wave][0], 32, 0, 0, lane);   // exp tile as A (16x32)
    v16h yb = load_b16(Yt, M, 0, j, lane);                 // onehot^T as B (32x16)
    acc = wmma16(ea, yb, acc);                             // acc[r][c] += sum_k e*1hot
  }
#pragma unroll
  for (int v = 0; v < 8; ++v) cacc[wave][rb + v][lane & 15] = acc[v];
  __syncthreads();
  if (wave == 0) {
    int cls = lane & 15;
    float sv[8];
#pragma unroll
    for (int v = 0; v < 8; ++v) {
      float s = 0.f;
      for (int w = 0; w < 8; ++w) s += cacc[w][rb + v][cls];
      sv[v] = s;
    }
#pragma unroll
    for (int v = 0; v < 8; ++v) {
      float x = (cls < 10) ? sv[v] : 0.f;     // denom = row-sum over real classes
#pragma unroll
      for (int m = 1; m < 16; m <<= 1) x += __shfl_xor(x, m, 32);
      if (cls < 10) {
        float o = sv[v] / x;
        out[(size_t)(r0 + rb + v) * 10 + cls] = fminf(fmaxf(o, 0.f), 1.f);
      }
    }
  }
}

extern "C" void kernel_launch(void* const* d_in, const int* in_sizes, int n_in,
                              void* d_out, int out_size, void* d_ws, size_t ws_size,
                              hipStream_t stream) {
  (void)n_in; (void)out_size; (void)ws_size;
  const float* x  = (const float*)d_in[0];
  const float* xn = (const float*)d_in[1];
  const int*   yn = (const int*)d_in[2];
  const float* W[3]  = {(const float*)d_in[3], (const float*)d_in[7],  (const float*)d_in[11]};
  const float* bv[3] = {(const float*)d_in[4], (const float*)d_in[8],  (const float*)d_in[12]};
  const float* ga[3] = {(const float*)d_in[5], (const float*)d_in[9],  (const float*)d_in[13]};
  const float* be[3] = {(const float*)d_in[6], (const float*)d_in[10], (const float*)d_in[14]};

  const int D0 = 1024, D1 = 512, D2 = 256, D3 = 128;
  const int N = in_sizes[0] / D0;   // 4096
  const int M = in_sizes[1] / D0;   // 8192
  const int dims[4] = {D0, D1, D2, D3};

  // workspace carve-out (256B aligned regions)
  char* wsp = (char*)d_ws;
  size_t off = 0;
  auto carve = [&](size_t bytes) -> char* {
    char* p = wsp + off;
    off += (bytes + 255) & ~(size_t)255;
    return p;
  };
  _Float16* XH  = (_Float16*)carve((size_t)N * D0 * 2);
  _Float16* XNH = (_Float16*)carve((size_t)M * D0 * 2);
  _Float16* H0  = (_Float16*)carve((size_t)M * D1 * 2);  // reused by both encodings
  _Float16* H1  = (_Float16*)carve((size_t)M * D2 * 2);
  _Float16* Qm  = (_Float16*)carve((size_t)N * D3 * 2);
  _Float16* Km  = (_Float16*)carve((size_t)M * D3 * 2);
  _Float16* WP  = (_Float16*)carve((size_t)D1 * D0 * 2); // max layer weight
  float*    BP  = (float*)carve((size_t)D1 * 4);
  float*    SV  = (float*)carve((size_t)D0 * 4);
  float*    TV  = (float*)carve((size_t)D0 * 4);
  float*    KK  = (float*)carve((size_t)M * 4);
  _Float16* YT  = (_Float16*)carve((size_t)16 * M * 2);
  float*    RM  = (float*)carve((size_t)N * 4);

  dim3 B(256);
  k_cvt_f16<<<(N * D0 + 255) / 256, B, 0, stream>>>(x, XH, N * D0);
  k_cvt_f16<<<(M * D0 + 255) / 256, B, 0, stream>>>(xn, XNH, M * D0);
  k_onehot_t<<<(16 * M + 255) / 256, B, 0, stream>>>(yn, YT, M);

  for (int enc = 0; enc < 2; ++enc) {
    int n = enc ? M : N;
    const float* srcF = enc ? xn : x;
    const _Float16* ins[3] = {enc ? XNH : XH, H0, H1};
    _Float16* outs[3] = {H0, H1, enc ? Km : Qm};
    for (int L = 0; L < 3; ++L) {
      int din = dims[L], dout = dims[L + 1];
      if (L == 0)
        k_colstats<float><<<din / 64, B, 0, stream>>>(srcF, ga[L], be[L], SV, TV, n, din);
      else
        k_colstats<_Float16><<<din / 64, B, 0, stream>>>(ins[L], ga[L], be[L], SV, TV, n, din);
      k_wfold<<<dout, B, 0, stream>>>(W[L], bv[L], SV, TV, WP, BP, din);
      k_gemm_tanh<<<n / 32, B, (size_t)32 * din * 2, stream>>>(ins[L], WP, BP, outs[L], din, dout);
    }
  }

  k_rownorm2<<<(M + 255) / 256, B, 0, stream>>>(Km, KK, M, D3);
  k_rowmax<<<N / 16, B, 0, stream>>>(Qm, Km, KK, RM, M);
  k_softmax_agg<<<N / 16, B, 0, stream>>>(Qm, Km, KK, YT, RM, (float*)d_out, M);
}
